// clique_gibbs_distr_61168924230075
// MI455X (gfx1250) — matrix operand order repaired and checked by reading, not compiled, not run on hardware
//
#include <hip/hip_runtime.h>

typedef __attribute__((ext_vector_type(16))) _Float16 v16h;
typedef __attribute__((ext_vector_type(8)))  _Float16 v8h;
typedef __attribute__((ext_vector_type(8)))  float    v8f;

#define BS  8
#define NA  128
#define NC  256
#define CS  3
#define NTY 3
#define NET 9

// f16 weight workspace layout (units: halves)
#define OFF_NW0 0          // 3 x [64][64]  ([h][d])
#define OFF_NW1 12288      // 3 x [64][64]  ([k][h])
#define OFF_NW2 24576      // 3 x [16][64]  ([z][k], rows 8..15 zero)
#define OFF_EW0 27648      // 9 x [64][64]  ([h][d])
#define OFF_EW1 64512      // 9 x [64][64]  ([k][h])
#define OFF_EW2 101376     // 9 x [64][64]  ([zz][k])
#define W_TOTAL 138240
#define WS_NF_BYTES 276480                      // nf: 8*128*8 f32
#define WS_EF_BYTES (276480 + 32768)            // ef: 6144*64 f32

// ---------------- WMMA fragment helpers (gfx1250 wave32) ----------------

__device__ __forceinline__ v16h frag_a(const _Float16* X, int lane, int kc) {
  // A 16x32 f16: lane m=lane&15 holds row m; halves 0..15 = K(16L..16L+15)
  // within the 32-wide K chunk.
  int m = lane & 15, L = lane >> 4;
  const _Float16* p0 = X + m * 64 + kc * 32 + 16 * L;
  v8h lo = *(const v8h*)p0;
  v8h hi = *(const v8h*)(p0 + 8);
  return __builtin_shufflevector(lo, hi, 0,1,2,3,4,5,6,7,8,9,10,11,12,13,14,15);
}

__device__ __forceinline__ v16h frag_b(const _Float16* W, int lane, int nc, int kc) {
  // B 32x16 f16: lane col=lane&15 holds column nc*16+col; 16 contiguous K
  // values starting at K = 16L within the 32-wide K chunk. W is [n][k], k-stride 64.
  int n = nc * 16 + (lane & 15), L = lane >> 4;
  return *(const v16h*)(W + n * 64 + kc * 32 + 16 * L);
}

__device__ __forceinline__ v8f wmma_f16(v16h a, v16h b, v8f c) {
  return __builtin_amdgcn_wmma_f32_16x16x32_f16(false, a, false, b, (short)0, c, false, false);
}

// One hidden layer: Y(16x64) = relu(X(16x64) @ W^T + bias). Y may alias X.
__device__ __forceinline__ void hidden_layer(const _Float16* X, _Float16* Y,
                                             const _Float16* W, const float* bias,
                                             int lane) {
  v16h a0 = frag_a(X, lane, 0);
  v16h a1 = frag_a(X, lane, 1);
  int col = lane & 15, rowbase = (lane >> 4) * 8;
#pragma unroll
  for (int nc = 0; nc < 4; ++nc) {
    v8f c = {};
    c = wmma_f16(a0, frag_b(W, lane, nc, 0), c);
    c = wmma_f16(a1, frag_b(W, lane, nc, 1), c);
    float bv = bias[nc * 16 + col];
#pragma unroll
    for (int r = 0; r < 8; ++r) {
      float v = c[r] + bv;
      v = v > 0.f ? v : 0.f;
      Y[(rowbase + r) * 64 + nc * 16 + col] = (_Float16)v;
    }
  }
}

// ---------------- weight transpose/convert to f16 ----------------

__global__ void prep_weights(const float* nW0, const float* nW1, const float* nW2,
                             const float* eW0, const float* eW1, const float* eW2,
                             _Float16* w) {
  int g = blockIdx.x * blockDim.x + threadIdx.x;
  if (g >= W_TOTAL) return;
  if (g < OFF_NW1) {                       // nW0 (t,d,h) -> [t][h][d]
    int j = g, t = j / 4096, r = j % 4096, h = r / 64, d = r % 64;
    w[g] = (_Float16)nW0[t * 4096 + d * 64 + h];
  } else if (g < OFF_NW2) {                // nW1 (t,h,k) -> [t][k][h]
    int j = g - OFF_NW1, t = j / 4096, r = j % 4096, k = r / 64, h = r % 64;
    w[g] = (_Float16)nW1[t * 4096 + h * 64 + k];
  } else if (g < OFF_EW0) {                // nW2 (t,k,8) -> [t][16][k] zero-padded
    int j = g - OFF_NW2, t = j / 1024, r = j % 1024, z = r / 64, k = r % 64;
    w[g] = (z < 8) ? (_Float16)nW2[t * 512 + k * 8 + z] : (_Float16)0.f;
  } else if (g < OFF_EW1) {                // eW0 (t,d,h) -> [t][h][d]
    int j = g - OFF_EW0, t = j / 4096, r = j % 4096, h = r / 64, d = r % 64;
    w[g] = (_Float16)eW0[t * 4096 + d * 64 + h];
  } else if (g < OFF_EW2) {                // eW1 (t,h,k) -> [t][k][h]
    int j = g - OFF_EW1, t = j / 4096, r = j % 4096, k = r / 64, h = r % 64;
    w[g] = (_Float16)eW1[t * 4096 + h * 64 + k];
  } else {                                 // eW2 (t,k,zz) -> [t][zz][k]
    int j = g - OFF_EW2, t = j / 4096, r = j % 4096, zz = r / 64, k = r % 64;
    w[g] = (_Float16)eW2[t * 4096 + k * 64 + zz];
  }
}

// ---------------- node factors: 64 tiles of 16 nodes, 1 wave each ----------------

__global__ __launch_bounds__(128)
void node_factor_kernel(const int* nt, const float* node_enc, const _Float16* wws,
                        const float* nb0, const float* nb1, const float* nb2,
                        float* nf_ws) {
  __shared__ __align__(32) _Float16 X0[4][16 * 64];
  __shared__ __align__(32) _Float16 X1[4][16 * 64];
  __shared__ int meta[4][16];
  int lane = threadIdx.x & 31, w = threadIdx.x >> 5;
  int T = blockIdx.x * 4 + w;                     // tile over bs*Na/16 = 64
  int m = lane & 15, L = lane >> 4;
  int g = T * 16 + m;                             // global node index
  if (L == 0) meta[w][m] = nt[g];
  const float* src = node_enc + (size_t)g * 64;
#pragma unroll
  for (int q = 0; q < 32; ++q)
    X0[w][m * 64 + L * 32 + q] = (_Float16)src[L * 32 + q];
  __builtin_amdgcn_wave_barrier();

  int rowbase = L * 8, col = lane & 15;
  int rowt[8];
#pragma unroll
  for (int r = 0; r < 8; ++r) rowt[r] = meta[w][rowbase + r];

  float acc[8];
#pragma unroll
  for (int r = 0; r < 8; ++r) acc[r] = 0.f;

  for (int tt = 0; tt < NTY; ++tt) {
    hidden_layer(X0[w], X1[w], wws + OFF_NW0 + tt * 4096, nb0 + tt * 64, lane);
    __builtin_amdgcn_wave_barrier();
    hidden_layer(X1[w], X1[w], wws + OFF_NW1 + tt * 4096, nb1 + tt * 64, lane);
    __builtin_amdgcn_wave_barrier();
    v16h a0 = frag_a(X1[w], lane, 0);
    v16h a1 = frag_a(X1[w], lane, 1);
    const _Float16* W2 = wws + OFF_NW2 + tt * 1024;     // [16][64]
    v8f c = {};
    c = wmma_f16(a0, frag_b(W2, lane, 0, 0), c);
    c = wmma_f16(a1, frag_b(W2, lane, 0, 1), c);
    float bv = (col < 8) ? nb2[tt * 8 + col] : 0.f;
#pragma unroll
    for (int r = 0; r < 8; ++r)
      acc[r] += (rowt[r] == tt) ? (c[r] + bv) : 0.f;
    __builtin_amdgcn_wave_barrier();
  }
  if (col < 8) {
#pragma unroll
    for (int r = 0; r < 8; ++r)
      nf_ws[(size_t)(T * 16 + rowbase + r) * 8 + col] = acc[r];
  }
}

// ------- edge factors: only the 6144 clique-referenced pairs, 384 tiles -------

__global__ __launch_bounds__(128)
void edge_factor_kernel(const int* nt, const float* edge_enc, const int* cni,
                        const _Float16* wws, const float* eb0, const float* eb1,
                        const float* eb2, float* ef_ws) {
  __shared__ __align__(32) _Float16 X0[4][16 * 64];
  __shared__ __align__(32) _Float16 X1[4][16 * 64];
  __shared__ int meta[4][16];
  int lane = threadIdx.x & 31, w = threadIdx.x >> 5;
  int T = blockIdx.x * 4 + w;                     // tile over 6144/16 = 384
  int m = lane & 15, L = lane >> 4;
  int p = T * 16 + m;                             // global pair index
  int b = p / (NC * CS);
  int rem = p % (NC * CS);
  int c = rem / CS, slot = rem % CS;
  const int sA[3] = {0, 0, 1}, sB[3] = {1, 2, 2};
  int I = cni[(b * NC + c) * CS + sA[slot]];
  int J = cni[(b * NC + c) * CS + sB[slot]];
  bool valid = (I >= 0) && (J >= 0);              // pad index -> zero factor
  int Iu = valid ? I : 0, Ju = valid ? J : 0;
  int t = valid ? (nt[b * NA + Ju] * 3 + nt[b * NA + Iu]) : -1;
  if (L == 0) meta[w][m] = t;
  const float* src = edge_enc + (((size_t)(b * NA + Iu)) * NA + Ju) * 64;
#pragma unroll
  for (int q = 0; q < 32; ++q)
    X0[w][m * 64 + L * 32 + q] = (_Float16)src[L * 32 + q];
  __builtin_amdgcn_wave_barrier();

  int rowbase = L * 8, col = lane & 15;
  int rowt[8];
#pragma unroll
  for (int r = 0; r < 8; ++r) rowt[r] = meta[w][rowbase + r];

  float acc[32];
#pragma unroll
  for (int q = 0; q < 32; ++q) acc[q] = 0.f;

  for (int tt = 0; tt < NET; ++tt) {
    hidden_layer(X0[w], X1[w], wws + OFF_EW0 + tt * 4096, eb0 + tt * 64, lane);
    __builtin_amdgcn_wave_barrier();
    hidden_layer(X1[w], X1[w], wws + OFF_EW1 + tt * 4096, eb1 + tt * 64, lane);
    __builtin_amdgcn_wave_barrier();
    v16h a0 = frag_a(X1[w], lane, 0);
    v16h a1 = frag_a(X1[w], lane, 1);
    const _Float16* W2 = wws + OFF_EW2 + tt * 4096;
#pragma unroll
    for (int nc = 0; nc < 4; ++nc) {
      v8f cacc = {};
      cacc = wmma_f16(a0, frag_b(W2, lane, nc, 0), cacc);
      cacc = wmma_f16(a1, frag_b(W2, lane, nc, 1), cacc);
      float bv = eb2[tt * 64 + nc * 16 + col];
#pragma unroll
      for (int r = 0; r < 8; ++r)
        acc[nc * 8 + r] += (rowt[r] == tt) ? (cacc[r] + bv) : 0.f;
    }
    __builtin_amdgcn_wave_barrier();
  }
#pragma unroll
  for (int nc = 0; nc < 4; ++nc)
#pragma unroll
    for (int r = 0; r < 8; ++r)
      ef_ws[(size_t)(T * 16 + rowbase + r) * 64 + nc * 16 + col] = acc[nc * 8 + r];
}

// ------- clique assembly: one block per (b, clique), 512 outputs -------

__global__ __launch_bounds__(512)
void assemble_kernel(const int* cni, const float* nf_ws, const float* ef_ws,
                     float* out) {
  __shared__ float nf_s[3][8];
  __shared__ float ef_s[3][64];
  int bc = blockIdx.x;                 // b*NC + c
  int tid = threadIdx.x;
  if (tid < 24) {
    int slot = tid >> 3, z = tid & 7;
    int I = cni[bc * 3 + slot];
    nf_s[slot][z] = (I >= 0) ? nf_ws[(size_t)((bc >> 8) * NA + I) * 8 + z] : 0.f;
  } else if (tid < 216) {
    int u = tid - 24, e = u >> 6, k = u & 63;
    ef_s[e][k] = ef_ws[(size_t)(bc * 3 + e) * 64 + k];
  }
  __syncthreads();
  int z0 = tid >> 6, z1 = (tid >> 3) & 7, z2 = tid & 7;
  out[(size_t)bc * 512 + tid] =
      nf_s[0][z0] + nf_s[1][z1] + nf_s[2][z2] +
      ef_s[0][z0 * 8 + z1] + ef_s[1][z0 * 8 + z2] + ef_s[2][z1 * 8 + z2];
}

__global__ void zmesh_kernel(float* out) {
  int g = blockIdx.x * blockDim.x + threadIdx.x;
  if (g >= 1536) return;
  int combo = g / 3, k = g % 3;
  int z0 = combo >> 6, z1 = (combo >> 3) & 7, z2 = combo & 7;
  out[1048576 + g] = (float)(k == 0 ? z0 : (k == 1 ? z1 : z2));
}

// ---------------- launcher ----------------

extern "C" void kernel_launch(void* const* d_in, const int* in_sizes, int n_in,
                              void* d_out, int out_size, void* d_ws, size_t ws_size,
                              hipStream_t stream) {
  (void)in_sizes; (void)n_in; (void)out_size; (void)ws_size;
  const int*   node_types = (const int*)d_in[0];
  const float* node_enc   = (const float*)d_in[1];
  const float* edge_enc   = (const float*)d_in[2];
  const int*   cni        = (const int*)d_in[4];   // clique_node_index
  const float* nW0 = (const float*)d_in[6];
  const float* nb0 = (const float*)d_in[7];
  const float* nW1 = (const float*)d_in[8];
  const float* nb1 = (const float*)d_in[9];
  const float* nW2 = (const float*)d_in[10];
  const float* nb2 = (const float*)d_in[11];
  const float* eW0 = (const float*)d_in[12];
  const float* eb0 = (const float*)d_in[13];
  const float* eW1 = (const float*)d_in[14];
  const float* eb1 = (const float*)d_in[15];
  const float* eW2 = (const float*)d_in[16];
  const float* eb2 = (const float*)d_in[17];

  _Float16* wws  = (_Float16*)d_ws;
  float* nf_ws   = (float*)((char*)d_ws + WS_NF_BYTES);
  float* ef_ws   = (float*)((char*)d_ws + WS_EF_BYTES);
  float* out     = (float*)d_out;

  prep_weights<<<(W_TOTAL + 255) / 256, 256, 0, stream>>>(nW0, nW1, nW2,
                                                          eW0, eW1, eW2, wws);
  node_factor_kernel<<<16, 128, 0, stream>>>(node_types, node_enc, wws,
                                             nb0, nb1, nb2, nf_ws);
  edge_factor_kernel<<<96, 128, 0, stream>>>(node_types, edge_enc, cni, wws,
                                             eb0, eb1, eb2, ef_ws);
  assemble_kernel<<<BS * NC, 512, 0, stream>>>(cni, nf_ws, ef_ws, out);
  zmesh_kernel<<<6, 256, 0, stream>>>(out);
}
